// Fp8PerBlockLinear_66984309949157
// MI455X (gfx1250) — 8-bit matrix operand order repaired and checked by reading, compiled, not run on hardware
//
#include <hip/hip_runtime.h>
#include <hip/hip_bf16.h>
#include <stdint.h>

typedef __attribute__((ext_vector_type(16))) int   v16i;
typedef __attribute__((ext_vector_type(8)))  float v8f;

#define MDIM 4096
#define KDIM 4096
#define NDIM 4096
#define GS   128
#define KGRP (KDIM / GS)
#define FP8M 448.0f
#define SEPS 1e-4f

static __device__ __forceinline__ float bf16_bits_to_f32(uint16_t h) {
  uint32_t u = ((uint32_t)h) << 16;
  return __builtin_bit_cast(float, u);
}

static __device__ __forceinline__ uint16_t f32_to_bf16_bits(float f) {
  uint32_t u = __builtin_bit_cast(uint32_t, f);
  u += 0x7FFFu + ((u >> 16) & 1u);
  return (uint16_t)(u >> 16);
}

static __device__ __forceinline__ uint32_t f32_to_e4m3_sw(float f) {
  uint32_t sign = (__builtin_bit_cast(uint32_t, f) >> 31) << 7;
  float af = fabsf(f);
  if (af < 0.015625f) {
    uint32_t m = (uint32_t)rintf(af * 512.0f);
    return sign | m;
  }
  uint32_t u = __builtin_bit_cast(uint32_t, af);
  u += 0x0007FFFFu + ((u >> 20) & 1u);
  int e = (int)(u >> 23) - 127 + 7;
  uint32_t m = (u >> 20) & 7u;
  if (e >= 16) return sign | 0x7Eu;
  return sign | ((uint32_t)e << 3) | m;
}

static __device__ __forceinline__ int lds_off(int row, int c16) {
  return row * 128 + ((c16 ^ ((row >> 1) & 7)) * 16);
}

static __device__ __forceinline__ void async_ld_b128(uint32_t ldsOff, const void* g) {
  asm volatile("global_load_async_to_lds_b128 %0, %1, off"
               :: "v"(ldsOff), "v"(g) : "memory");
}
static __device__ __forceinline__ void async_ld_b64(uint32_t ldsOff, const void* g) {
  asm volatile("global_load_async_to_lds_b64 %0, %1, off"
               :: "v"(ldsOff), "v"(g) : "memory");
}
static __device__ __forceinline__ void wait_async0() {
#if __has_builtin(__builtin_amdgcn_s_wait_asynccnt)
  __builtin_amdgcn_s_wait_asynccnt(0);
#else
  asm volatile("s_wait_asynccnt 0x0" ::: "memory");
#endif
}

__global__ __launch_bounds__(256) void quant_x_fp8(
    const uint16_t* __restrict__ xbf,
    uint8_t* __restrict__ xq,
    float*   __restrict__ xsT)
{
  const int  lane = threadIdx.x & 31;
  const int  wv   = threadIdx.x >> 5;
  const long grp  = (long)blockIdx.x * 8 + wv;
  const long base = grp * GS;

  const ushort4 u = *(const ushort4*)(xbf + base + lane * 4);
  float f0 = bf16_bits_to_f32(u.x);
  float f1 = bf16_bits_to_f32(u.y);
  float f2 = bf16_bits_to_f32(u.z);
  float f3 = bf16_bits_to_f32(u.w);

  float amax = fmaxf(fmaxf(fabsf(f0), fabsf(f1)), fmaxf(fabsf(f2), fabsf(f3)));
#pragma unroll
  for (int off = 16; off >= 1; off >>= 1)
    amax = fmaxf(amax, __shfl_xor(amax, off, 32));

  const float s   = fmaxf(amax * (1.0f / FP8M), SEPS);
  const float inv = 1.0f / s;
  f0 = fminf(fmaxf(f0 * inv, -FP8M), FP8M);
  f1 = fminf(fmaxf(f1 * inv, -FP8M), FP8M);
  f2 = fminf(fmaxf(f2 * inv, -FP8M), FP8M);
  f3 = fminf(fmaxf(f3 * inv, -FP8M), FP8M);

  int pk;
#if __has_builtin(__builtin_amdgcn_cvt_pk_fp8_f32)
  pk = __builtin_amdgcn_cvt_pk_fp8_f32(f0, f1, 0, false);
  pk = __builtin_amdgcn_cvt_pk_fp8_f32(f2, f3, pk, true);
#else
  pk = (int)(f32_to_e4m3_sw(f0)        | (f32_to_e4m3_sw(f1) << 8) |
             (f32_to_e4m3_sw(f2) << 16) | (f32_to_e4m3_sw(f3) << 24));
#endif
  *(int*)(xq + base + lane * 4) = pk;
  if (lane == 0) {
    const long m  = grp >> 5;
    const long kg = grp & (KGRP - 1);
    xsT[kg * MDIM + m] = s;
  }
}

__global__ __launch_bounds__(256) void gemm_fp8_wmma(
    const uint8_t*  __restrict__ xq,
    const float*    __restrict__ xsT,
    const uint8_t*  __restrict__ wq,
    const float*    __restrict__ wsc,
    const uint16_t* __restrict__ bias,
    uint16_t*       __restrict__ out)
{
  __shared__ uint8_t smem[65536];

  const int t    = threadIdx.x;
  const int lane = t & 31;
  const int wv   = t >> 5;
  const int hl   = lane >> 4;
  const int l15  = lane & 15;
  const int wm   = wv & 3;
  const int wn   = wv >> 2;
  const int mBlk = blockIdx.x * 128;
  const int nBlk = blockIdx.y * 128;
  const int m0   = mBlk + wm * 32;
  const int n0   = nBlk + wn * 64;

  const uint32_t sbase = (uint32_t)(uintptr_t)smem;

  const uint8_t* gA[8];
  uint32_t aLds[8];
#pragma unroll
  for (int j = 0; j < 8; ++j) {
    const int c   = t + j * 256;
    const int row = c >> 4;
    const int g   = c & 7;
    const int h   = (c >> 3) & 1;
    gA[j] = xq + (size_t)(mBlk + row) * KDIM + g * 16 + h * 8;
    const int cA = (g & 1) * 4 + (g >> 1);
    aLds[j] = (uint32_t)(row * 128 + ((cA ^ ((row >> 1) & 7)) * 16) + h * 8);
  }
  const uint8_t* gB[4];
  uint32_t bLds[4];
#pragma unroll
  for (int j = 0; j < 4; ++j) {
    const int c   = t + j * 256;
    const int row = c >> 3;
    const int c16 = c & 7;
    gB[j] = wq + (size_t)(nBlk + row) * KDIM + c16 * 16;
    bLds[j] = (uint32_t)lds_off(row, c16);
  }

  v8f acc[2][4];
#pragma unroll
  for (int a = 0; a < 2; ++a)
#pragma unroll
    for (int b = 0; b < 4; ++b) acc[a][b] = {};
  const v8f czero = {};

  const float* swp = wsc + (size_t)blockIdx.y * KGRP;

#pragma unroll
  for (int j = 0; j < 8; ++j) async_ld_b64(sbase + aLds[j], gA[j]);
#pragma unroll
  for (int j = 0; j < 4; ++j) async_ld_b128(sbase + 32768u + bLds[j], gB[j]);
  wait_async0();
  __syncthreads();

  for (int kg = 0; kg < KGRP; ++kg) {
    const int cur = kg & 1;

    if (kg + 1 < KGRP) {
      const int ko = (kg + 1) * GS;
      const uint32_t abuf = sbase + (uint32_t)((cur ^ 1) * 16384);
      const uint32_t bbuf = sbase + 32768u + (uint32_t)((cur ^ 1) * 16384);
#pragma unroll
      for (int j = 0; j < 8; ++j) async_ld_b64(abuf + aLds[j], gA[j] + ko);
#pragma unroll
      for (int j = 0; j < 4; ++j) async_ld_b128(bbuf + bLds[j], gB[j] + ko);
    }

    const uint8_t* la = smem + cur * 16384;
    const uint8_t* lb = smem + 32768 + cur * 16384;

    v16i a0, a1;
#pragma unroll
    for (int q = 0; q < 4; ++q) {
      const int r0 = wm * 32 + l15;
      const int r1 = r0 + 16;
      const int cA = hl * 4 + q;
      int4 t0 = *(const int4*)(la + r0 * 128 + ((cA ^ ((r0 >> 1) & 7)) * 16));
      int4 t1 = *(const int4*)(la + r1 * 128 + ((cA ^ ((r1 >> 1) & 7)) * 16));
      a0[4 * q] = t0.x; a0[4 * q + 1] = t0.y; a0[4 * q + 2] = t0.z; a0[4 * q + 3] = t0.w;
      a1[4 * q] = t1.x; a1[4 * q + 1] = t1.y; a1[4 * q + 2] = t1.z; a1[4 * q + 3] = t1.w;
    }

    const float  sw  = swp[kg];
    const float* xr0 = xsT + (size_t)kg * MDIM + (m0 + 8 * hl);
    const float* xr1 = xr0 + 16;
    float4 s0a = *(const float4*)(xr0);
    float4 s0b = *(const float4*)(xr0 + 4);
    float4 s1a = *(const float4*)(xr1);
    float4 s1b = *(const float4*)(xr1 + 4);
    float sv0[8] = {s0a.x * sw, s0a.y * sw, s0a.z * sw, s0a.w * sw,
                    s0b.x * sw, s0b.y * sw, s0b.z * sw, s0b.w * sw};
    float sv1[8] = {s1a.x * sw, s1a.y * sw, s1a.z * sw, s1a.w * sw,
                    s1b.x * sw, s1b.y * sw, s1b.z * sw, s1b.w * sw};

#pragma unroll
    for (int ns = 0; ns < 4; ++ns) {
      const int rn = wn * 64 + ns * 16 + l15;
      v16i b;
#pragma unroll
      for (int q = 0; q < 4; ++q) {
        int4 tb = *(const int4*)(lb + lds_off(rn, 2 * q + hl));
        b[4 * q] = tb.x; b[4 * q + 1] = tb.y; b[4 * q + 2] = tb.z; b[4 * q + 3] = tb.w;
      }
      v8f p0 = __builtin_amdgcn_wmma_f32_16x16x128_fp8_fp8(a0, b, (short)0, czero, false, false);
      v8f p1 = __builtin_amdgcn_wmma_f32_16x16x128_fp8_fp8(a1, b, (short)0, czero, false, false);
#pragma unroll
      for (int i = 0; i < 8; ++i) {
        acc[0][ns][i] += sv0[i] * p0[i];
        acc[1][ns][i] += sv1[i] * p1[i];
      }
    }

    wait_async0();
    __syncthreads();
  }

#pragma unroll
  for (int ns = 0; ns < 4; ++ns) {
    const int n  = n0 + ns * 16 + l15;
    const float bf = bf16_bits_to_f32(bias[n]);
#pragma unroll
    for (int ms = 0; ms < 2; ++ms) {
#pragma unroll
      for (int i = 0; i < 8; ++i) {
        const int m = m0 + ms * 16 + 8 * hl + i;
        out[(size_t)m * NDIM + n] = f32_to_bf16_bits(acc[ms][ns][i] + bf);
      }
    }
  }
}

extern "C" void kernel_launch(void* const* d_in, const int* in_sizes, int n_in,
                              void* d_out, int out_size, void* d_ws, size_t ws_size,
                              hipStream_t stream) {
  (void)in_sizes; (void)n_in; (void)out_size; (void)ws_size;
  const uint16_t* x      = (const uint16_t*)d_in[0];
  const uint8_t*  weight = (const uint8_t*)d_in[1];
  const float*    wscale = (const float*)d_in[2];
  const uint16_t* bias   = (const uint16_t*)d_in[3];
  uint16_t*       out    = (uint16_t*)d_out;

  uint8_t* xq  = (uint8_t*)d_ws;
  float*   xsT = (float*)((uint8_t*)d_ws + (size_t)MDIM * KDIM);

  quant_x_fp8<<<(MDIM * KGRP) / 8, 256, 0, stream>>>(x, xq, xsT);

  dim3 grid(MDIM / 128, NDIM / 128);
  gemm_fp8_wmma<<<grid, 256, 0, stream>>>(xq, xsT, weight, wscale, bias, out);
}
